// MoETransformerEncoderLayer_15968688406695
// MI455X (gfx1250) — compile-verified, matrix operand order checked by reading, not run on hardware
//
#include <hip/hip_runtime.h>
#include <hip/hip_bf16.h>

typedef __attribute__((ext_vector_type(16))) __bf16 v16bf;
typedef __attribute__((ext_vector_type(8)))  float  v8f;

#define S_ 1024
#define B_ 4
#define D_ 1024
#define H_ 16
#define NE_ 8
#define DH_ 4096
#define NT_ 4096      // tokens = S*B
#define CAP_ 4096     // per-expert token capacity

// ---------------- fragment loader -------------------------------------------
// LDS tile stored as [outdim][K] with stride STA; A/B 16x32 bf16 fragment:
// lane l: row=l%16, half=l/16; VGPR v holds K pair at (v/4)*16+half*8+(v%4)*2.
__device__ __forceinline__ v16bf load_frag(const __bf16* base, int stride,
                                           int row, int half) {
  v16bf f;
  const __bf16* rp = base + row * stride;
#pragma unroll
  for (int v = 0; v < 8; ++v) {
    const int k0 = ((v >> 2) << 4) + (half << 3) + ((v & 3) << 1);
    union { unsigned u; __bf16 h[2]; } cv;
    cv.u = *reinterpret_cast<const unsigned*>(rp + k0);
    f[2 * v]     = cv.h[0];
    f[2 * v + 1] = cv.h[1];
  }
  return f;
}

__device__ __forceinline__ v8f wmma_bf16(v16bf a, v16bf b, v8f c) {
  return __builtin_amdgcn_wmma_f32_16x16x32_bf16(false, a, false, b,
                                                 (short)0, c, false, false);
}

// ---------------- async double-buffered GEMM core ---------------------------
// C[128x64] = A[M][K] (bf16, optional row-gather) @ B[N][K]^T (bf16).
// 256 thr = 8 waves; wave w owns rows w*16..w*16+15, all 64 cols (4 C tiles).
// Tiles staged with global_load_async_to_lds_b128, double buffered,
// pipelined with s_wait_asynccnt.
#define BM2 128
#define BN2 64
#define STA 72   // padded stride (elems): 144B rows keep 16B chunk alignment

template <bool GATHER>
__device__ void gemm_core2(const __bf16* __restrict__ A, long lda,
                           const int* __restrict__ rowmap, int rshift,
                           const __bf16* __restrict__ Bm, long ldb,
                           int Mvalid, int K, int row0, int n0, v8f* acc) {
  __shared__ __bf16 As[2][BM2][STA];
  __shared__ __bf16 Bs[2][BN2][STA];
  const int tid  = threadIdx.x;
  const int wave = tid >> 5, lane = tid & 31;
  const int half = lane >> 4, lm = lane & 15;
  const int lrow = lane >> 3;          // 0..3 : row inside 4-row chunk
  const int lcol = (lane & 7) * 16;    // byte offset inside 128B row

  const unsigned asbase = (unsigned)(unsigned long)&As[0][0][0];
  const unsigned bsbase = (unsigned)(unsigned long)&Bs[0][0][0];

  auto issue = [&](int buf, int k0) {
    // A: wave covers its own 16 rows (4 chunks of 4 rows x 128B)
#pragma unroll
    for (int ch = 0; ch < 4; ++ch) {
      const int r  = wave * 16 + ch * 4 + lrow;
      const int gr = row0 + r;
      long arow;
      if (GATHER) arow = (gr < Mvalid) ? (long)(rowmap[gr] >> rshift) : 0L;
      else        arow = (long)gr;
      const char* gp = (const char*)(A + arow * lda + k0) + lcol;
      const unsigned lo = asbase + (unsigned)((buf * BM2 + r) * STA * 2) + lcol;
      asm volatile("global_load_async_to_lds_b128 %0, %1, off"
                   :: "v"(lo), "v"(gp) : "memory");
    }
    // B: wave covers rows 8w..8w+7 (2 chunks)
#pragma unroll
    for (int ch = 0; ch < 2; ++ch) {
      const int n = wave * 8 + ch * 4 + lrow;
      const char* gp = (const char*)(Bm + (long)(n0 + n) * ldb + k0) + lcol;
      const unsigned lo = bsbase + (unsigned)((buf * BN2 + n) * STA * 2) + lcol;
      asm volatile("global_load_async_to_lds_b128 %0, %1, off"
                   :: "v"(lo), "v"(gp) : "memory");
    }
  };

  auto compute = [&](int buf) {
#pragma unroll
    for (int ks = 0; ks < 64; ks += 32) {
      v16bf a = load_frag(&As[buf][wave * 16][ks], STA, lm, half);
#pragma unroll
      for (int t = 0; t < 4; ++t) {
        v16bf b = load_frag(&Bs[buf][t * 16][ks], STA, lm, half);
        acc[t] = wmma_bf16(a, b, acc[t]);
      }
    }
  };

  const int nk = K >> 6;
  issue(0, 0);
  for (int kt = 0; kt < nk; ++kt) {
    const int cur = kt & 1;
    if (kt + 1 < nk) {
      issue(cur ^ 1, (kt + 1) << 6);
      asm volatile("s_wait_asynccnt 0x6" ::: "memory");  // cur tile landed
    } else {
      asm volatile("s_wait_asynccnt 0x0" ::: "memory");
    }
    __syncthreads();   // everyone's copies for `cur` are visible
    compute(cur);
    __syncthreads();   // done reading `cur` before it is refilled
  }
}

// ---------------- conversion / transpose kernels ----------------------------
__global__ __launch_bounds__(256) void k_cvt_bf16(const float* __restrict__ in,
                                                  __bf16* __restrict__ out, int n) {
  int i = blockIdx.x * 256 + threadIdx.x;
  if (i < n) out[i] = (__bf16)in[i];
}

// out[c*R + r] = (bf16) in[r*C + c] ; per-expert slab via blockIdx.z
__global__ __launch_bounds__(256) void k_tr_cvt(const float* __restrict__ in,
                                                __bf16* __restrict__ out,
                                                int R, int C) {
  __shared__ float tile[32][33];
  const long base = (long)blockIdx.z * R * C;
  const int r0 = blockIdx.y * 32, c0 = blockIdx.x * 32;
  const int tx = threadIdx.x & 31, ty = threadIdx.x >> 5;
#pragma unroll
  for (int i = 0; i < 32; i += 8)
    tile[ty + i][tx] = in[base + (long)(r0 + ty + i) * C + c0 + tx];
  __syncthreads();
#pragma unroll
  for (int i = 0; i < 32; i += 8)
    out[base + (long)(c0 + ty + i) * R + r0 + tx] = (__bf16)tile[tx][ty + i];
}

// ---------------- GEMM kernels ----------------------------------------------
// qkv = src @ Wqkv^T + bqkv (q cols pre-scaled by 1/sqrt(64))
__global__ __launch_bounds__(256) void k_gemm_qkv(const __bf16* __restrict__ A,
                                                  const __bf16* __restrict__ W,
                                                  const float* __restrict__ bias,
                                                  __bf16* __restrict__ out) {
  const int row0 = blockIdx.y * BM2, n0 = blockIdx.x * BN2;
  v8f acc[4] = {};
  gemm_core2<false>(A, D_, nullptr, 0, W, D_, NT_, D_, row0, n0, acc);
  const int wave = threadIdx.x >> 5, lane = threadIdx.x & 31;
  const int half = lane >> 4, lm = lane & 15;
#pragma unroll
  for (int t = 0; t < 4; ++t) {
    const int n = n0 + t * 16 + lm;
    const float sc = (n < D_) ? 0.125f : 1.0f;
    const float bi = bias[n];
#pragma unroll
    for (int r = 0; r < 8; ++r) {
      const int m = row0 + wave * 16 + half * 8 + r;
      out[(long)m * (3 * D_) + n] = (__bf16)((acc[t][r] + bi) * sc);
    }
  }
}

// attnproj = o @ Wo^T + bo (fp32 out)
__global__ __launch_bounds__(256) void k_gemm_wo(const __bf16* __restrict__ A,
                                                 const __bf16* __restrict__ W,
                                                 const float* __restrict__ bias,
                                                 float* __restrict__ out) {
  const int row0 = blockIdx.y * BM2, n0 = blockIdx.x * BN2;
  v8f acc[4] = {};
  gemm_core2<false>(A, D_, nullptr, 0, W, D_, NT_, D_, row0, n0, acc);
  const int wave = threadIdx.x >> 5, lane = threadIdx.x & 31;
  const int half = lane >> 4, lm = lane & 15;
#pragma unroll
  for (int t = 0; t < 4; ++t) {
    const int n = n0 + t * 16 + lm;
    const float bi = bias[n];
#pragma unroll
    for (int r = 0; r < 8; ++r) {
      const int m = row0 + wave * 16 + half * 8 + r;
      out[(long)m * D_ + n] = acc[t][r] + bi;
    }
  }
}

// H[token*2+k] = gelu(x[tok] @ W1[e] + b1[e])     (W1t: [DH][D] bf16)
__global__ __launch_bounds__(256) void k_moe_gemm1(const __bf16* __restrict__ xbf,
                                                   const __bf16* __restrict__ W1t,
                                                   const float* __restrict__ b1,
                                                   const int* __restrict__ cnt,
                                                   const int* __restrict__ elist,
                                                   __bf16* __restrict__ Hbuf) {
  const int e = blockIdx.z;
  const int count = cnt[e];
  const int row0 = blockIdx.y * BM2;
  if (row0 >= count) return;
  const int n0 = blockIdx.x * BN2;
  const int* rmap = elist + e * CAP_;
  v8f acc[4] = {};
  gemm_core2<true>(xbf, D_, rmap, 1, W1t + (long)e * D_ * DH_, D_,
                   count, D_, row0, n0, acc);
  const int wave = threadIdx.x >> 5, lane = threadIdx.x & 31;
  const int half = lane >> 4, lm = lane & 15;
#pragma unroll
  for (int t = 0; t < 4; ++t) {
    const int n = n0 + t * 16 + lm;
    const float bi = b1[e * DH_ + n];
#pragma unroll
    for (int r = 0; r < 8; ++r) {
      const int mi = row0 + wave * 16 + half * 8 + r;
      if (mi < count) {
        const int packed = rmap[mi];
        const float v = acc[t][r] + bi;
        const float gl = 0.5f * v * (1.f + erff(v * 0.70710678f));
        Hbuf[(long)packed * DH_ + n] = (__bf16)gl;
      }
    }
  }
}

// ff[token] += gate * (H @ W2[e] + b2[e])         (W2t: [D][DH] bf16)
__global__ __launch_bounds__(256) void k_moe_gemm2(const __bf16* __restrict__ Hbuf,
                                                   const __bf16* __restrict__ W2t,
                                                   const float* __restrict__ b2,
                                                   const int* __restrict__ cnt,
                                                   const int* __restrict__ elist,
                                                   const float* __restrict__ agate,
                                                   float* __restrict__ ff) {
  const int e = blockIdx.z;
  const int count = cnt[e];
  const int row0 = blockIdx.y * BM2;
  if (row0 >= count) return;
  const int n0 = blockIdx.x * BN2;
  const int* rmap = elist + e * CAP_;
  v8f acc[4] = {};
  gemm_core2<true>(Hbuf, DH_, rmap, 0, W2t + (long)e * DH_ * D_, DH_,
                   count, DH_, row0, n0, acc);
  const int wave = threadIdx.x >> 5, lane = threadIdx.x & 31;
  const int half = lane >> 4, lm = lane & 15;
#pragma unroll
  for (int t = 0; t < 4; ++t) {
    const int n = n0 + t * 16 + lm;
    const float bi = b2[e * D_ + n];
#pragma unroll
    for (int r = 0; r < 8; ++r) {
      const int mi = row0 + wave * 16 + half * 8 + r;
      if (mi < count) {
        const int packed = rmap[mi];
        const int token = packed >> 1;
        atomicAdd(&ff[(long)token * D_ + n], (acc[t][r] + bi) * agate[packed]);
      }
    }
  }
}

// ---------------- flash attention --------------------------------------------
__global__ __launch_bounds__(128) void k_flash_attn(const __bf16* __restrict__ qkv,
                                                    __bf16* __restrict__ obuf) {
  const int i0 = blockIdx.x * 64;
  const int bh = blockIdx.y;
  const int b = bh >> 4, h = bh & 15;
  __shared__ __bf16 Qs[64][68];
  __shared__ __bf16 Ks[64][68];
  __shared__ __bf16 Vt[64][68];
  __shared__ __bf16 Ps[64][68];
  const int tid = threadIdx.x, wave = tid >> 5, lane = tid & 31;
  const int half = lane >> 4, lm = lane & 15;

  for (int i = tid; i < 64 * 64; i += 128) {
    const int r = i >> 6, c = i & 63;
    const long tok = (long)(i0 + r) * B_ + b;
    Qs[r][c] = qkv[tok * (3 * D_) + h * 64 + c];
  }
  __syncthreads();
  v16bf qlo = load_frag(&Qs[wave * 16][0], 68, lm, half);
  v16bf qhi = load_frag(&Qs[wave * 16][32], 68, lm, half);

  float mrow[8], lrow[8];
  v8f oacc[4] = {};
#pragma unroll
  for (int r = 0; r < 8; ++r) { mrow[r] = -1e30f; lrow[r] = 0.f; }

  for (int j = 0; j < S_ / 64; ++j) {
    for (int i = tid; i < 64 * 64; i += 128) {
      const int r = i >> 6, c = i & 63;
      const long tok = (long)(j * 64 + r) * B_ + b;
      Ks[r][c] = qkv[tok * (3 * D_) + D_ + h * 64 + c];
      Vt[c][r] = qkv[tok * (3 * D_) + 2 * D_ + h * 64 + c];
    }
    __syncthreads();

    v8f sf[4] = {};
#pragma unroll
    for (int t = 0; t < 4; ++t) {
      v16bf blo = load_frag(&Ks[t * 16][0], 68, lm, half);
      v16bf bhi = load_frag(&Ks[t * 16][32], 68, lm, half);
      sf[t] = wmma_bf16(qlo, blo, sf[t]);
      sf[t] = wmma_bf16(qhi, bhi, sf[t]);
    }
#pragma unroll
    for (int r = 0; r < 8; ++r) {
      float mx = fmaxf(fmaxf(sf[0][r], sf[1][r]), fmaxf(sf[2][r], sf[3][r]));
#pragma unroll
      for (int o = 8; o >= 1; o >>= 1) mx = fmaxf(mx, __shfl_xor(mx, o));
      const float mnew  = fmaxf(mrow[r], mx);
      const float alpha = __expf(mrow[r] - mnew);
      float rs = 0.f;
#pragma unroll
      for (int t = 0; t < 4; ++t) {
        const float p = __expf(sf[t][r] - mnew);
        sf[t][r] = p;
        rs += p;
      }
#pragma unroll
      for (int o = 8; o >= 1; o >>= 1) rs += __shfl_xor(rs, o);
      lrow[r] = lrow[r] * alpha + rs;
      mrow[r] = mnew;
#pragma unroll
      for (int t = 0; t < 4; ++t) oacc[t][r] *= alpha;
    }
#pragma unroll
    for (int t = 0; t < 4; ++t)
#pragma unroll
      for (int r = 0; r < 8; ++r)
        Ps[wave * 16 + half * 8 + r][t * 16 + lm] = (__bf16)sf[t][r];
    __syncthreads();

    v16bf plo = load_frag(&Ps[wave * 16][0], 68, lm, half);
    v16bf phi = load_frag(&Ps[wave * 16][32], 68, lm, half);
#pragma unroll
    for (int t = 0; t < 4; ++t) {
      v16bf vlo = load_frag(&Vt[t * 16][0], 68, lm, half);
      v16bf vhi = load_frag(&Vt[t * 16][32], 68, lm, half);
      oacc[t] = wmma_bf16(plo, vlo, oacc[t]);
      oacc[t] = wmma_bf16(phi, vhi, oacc[t]);
    }
    __syncthreads();
  }
#pragma unroll
  for (int r = 0; r < 8; ++r) {
    const int m = wave * 16 + half * 8 + r;
    const long tok = (long)(i0 + m) * B_ + b;
    const float inv = 1.f / lrow[r];
#pragma unroll
    for (int t = 0; t < 4; ++t)
      obuf[tok * D_ + h * 64 + t * 16 + lm] = (__bf16)(oacc[t][r] * inv);
  }
}

// ---------------- layernorm / gating -----------------------------------------
__global__ __launch_bounds__(256) void k_ln(const float* __restrict__ a,
                                            const float* __restrict__ res,
                                            const float* __restrict__ g,
                                            const float* __restrict__ be,
                                            float* __restrict__ outf,
                                            __bf16* __restrict__ outb) {
  const int row = blockIdx.x, tid = threadIdx.x;
  const int lane = tid & 31, wave = tid >> 5;
  __shared__ float red[8];
  const long base = (long)row * D_;
  float v[4];
  float s = 0.f;
#pragma unroll
  for (int i = 0; i < 4; ++i) {
    const int c = tid + i * 256;
    v[i] = a[base + c] + res[base + c];
    s += v[i];
  }
#pragma unroll
  for (int o = 16; o >= 1; o >>= 1) s += __shfl_xor(s, o);
  if (lane == 0) red[wave] = s;
  __syncthreads();
  float tot = 0.f;
#pragma unroll
  for (int w = 0; w < 8; ++w) tot += red[w];
  const float mean = tot * (1.f / D_);
  float vs = 0.f;
#pragma unroll
  for (int i = 0; i < 4; ++i) { const float d = v[i] - mean; vs += d * d; }
#pragma unroll
  for (int o = 16; o >= 1; o >>= 1) vs += __shfl_xor(vs, o);
  __syncthreads();
  if (lane == 0) red[wave] = vs;
  __syncthreads();
  float vtot = 0.f;
#pragma unroll
  for (int w = 0; w < 8; ++w) vtot += red[w];
  const float rstd = rsqrtf(vtot * (1.f / D_) + 1e-5f);
#pragma unroll
  for (int i = 0; i < 4; ++i) {
    const int c = tid + i * 256;
    const float y = (v[i] - mean) * rstd * g[c] + be[c];
    outf[base + c] = y;
    if (outb) outb[base + c] = (__bf16)y;
  }
}

__global__ __launch_bounds__(256) void k_gate(const float* __restrict__ x,
                                              const float* __restrict__ wg,
                                              int* __restrict__ cnt,
                                              int* __restrict__ elist,
                                              float* __restrict__ agate) {
  const int wave = threadIdx.x >> 5, lane = threadIdx.x & 31;
  const int token = blockIdx.x * 8 + wave;
  float acc[NE_] = {};
  for (int d = lane; d < D_; d += 32) {
    const float xv = x[(long)token * D_ + d];
#pragma unroll
    for (int e = 0; e < NE_; ++e) acc[e] += xv * wg[d * NE_ + e];
  }
#pragma unroll
  for (int e = 0; e < NE_; ++e)
#pragma unroll
    for (int o = 16; o >= 1; o >>= 1) acc[e] += __shfl_xor(acc[e], o);
  if (lane == 0) {
    int e1 = 0; float l1 = acc[0];
#pragma unroll
    for (int e = 1; e < NE_; ++e) if (acc[e] > l1) { l1 = acc[e]; e1 = e; }
    int e2 = (e1 == 0) ? 1 : 0; float l2 = acc[e2];
#pragma unroll
    for (int e = 0; e < NE_; ++e)
      if (e != e1 && acc[e] > l2) { l2 = acc[e]; e2 = e; }
    const float g1 = 1.f / (1.f + __expf(l2 - l1));
    const float g2 = 1.f - g1;
    const int s1 = atomicAdd(&cnt[e1], 1);
    elist[e1 * CAP_ + s1] = token * 2 + 0;
    agate[token * 2 + 0] = g1;
    const int s2 = atomicAdd(&cnt[e2], 1);
    elist[e2 * CAP_ + s2] = token * 2 + 1;
    agate[token * 2 + 1] = g2;
  }
}

// ---------------- host wiring ------------------------------------------------
extern "C" void kernel_launch(void* const* d_in, const int* in_sizes, int n_in,
                              void* d_out, int out_size, void* d_ws, size_t ws_size,
                              hipStream_t stream) {
  const float* src    = (const float*)d_in[0];
  const float* Wqkv   = (const float*)d_in[1];
  const float* bqkv   = (const float*)d_in[2];
  const float* Wo     = (const float*)d_in[3];
  const float* bo     = (const float*)d_in[4];
  const float* ln1_g  = (const float*)d_in[5];
  const float* ln1_b  = (const float*)d_in[6];
  const float* ln2_g  = (const float*)d_in[7];
  const float* ln2_b  = (const float*)d_in[8];
  const float* w_gate = (const float*)d_in[9];
  const float* W1     = (const float*)d_in[10];
  const float* b1     = (const float*)d_in[11];
  const float* W2     = (const float*)d_in[12];
  const float* b2     = (const float*)d_in[13];
  (void)in_sizes; (void)n_in; (void)out_size; (void)ws_size;

  char* ws = (char*)d_ws;
  size_t off = 0;
  auto alloc = [&](size_t bytes) { char* p = ws + off; off = (off + bytes + 255) & ~(size_t)255; return p; };
  __bf16* src_bf  = (__bf16*)alloc((size_t)NT_ * D_ * 2);
  __bf16* wqkv_bf = (__bf16*)alloc((size_t)3 * D_ * D_ * 2);
  __bf16* wo_bf   = (__bf16*)alloc((size_t)D_ * D_ * 2);
  __bf16* w1t_bf  = (__bf16*)alloc((size_t)NE_ * D_ * DH_ * 2);
  __bf16* w2t_bf  = (__bf16*)alloc((size_t)NE_ * DH_ * D_ * 2);
  __bf16* qkv_bf  = (__bf16*)alloc((size_t)NT_ * 3 * D_ * 2);
  __bf16* o_bf    = (__bf16*)alloc((size_t)NT_ * D_ * 2);
  float*  aproj   = (float*)alloc((size_t)NT_ * D_ * 4);
  float*  x_f     = (float*)alloc((size_t)NT_ * D_ * 4);
  __bf16* x_bf    = (__bf16*)alloc((size_t)NT_ * D_ * 2);
  int*    cnt     = (int*)alloc(256);
  int*    elist   = (int*)alloc((size_t)NE_ * CAP_ * 4);
  float*  agate   = (float*)alloc((size_t)2 * NT_ * 4);
  __bf16* Hbuf    = (__bf16*)alloc((size_t)2 * NT_ * DH_ * 2);
  float*  ff      = (float*)alloc((size_t)NT_ * D_ * 4);

  hipMemsetAsync(cnt, 0, 256, stream);
  hipMemsetAsync(ff, 0, (size_t)NT_ * D_ * 4, stream);

  // precision staging: everything the WMMA pipes touch goes bf16 once
  k_cvt_bf16<<<(NT_ * D_ + 255) / 256, 256, 0, stream>>>(src, src_bf, NT_ * D_);
  k_cvt_bf16<<<(3 * D_ * D_ + 255) / 256, 256, 0, stream>>>(Wqkv, wqkv_bf, 3 * D_ * D_);
  k_cvt_bf16<<<(D_ * D_ + 255) / 256, 256, 0, stream>>>(Wo, wo_bf, D_ * D_);
  k_tr_cvt<<<dim3(DH_ / 32, D_ / 32, NE_), 256, 0, stream>>>(W1, w1t_bf, D_, DH_);
  k_tr_cvt<<<dim3(D_ / 32, DH_ / 32, NE_), 256, 0, stream>>>(W2, w2t_bf, DH_, D_);

  // attention
  k_gemm_qkv<<<dim3(3 * D_ / BN2, NT_ / BM2), 256, 0, stream>>>(src_bf, wqkv_bf, bqkv, qkv_bf);
  k_flash_attn<<<dim3(S_ / 64, B_ * H_), 128, 0, stream>>>(qkv_bf, o_bf);
  k_gemm_wo<<<dim3(D_ / BN2, NT_ / BM2), 256, 0, stream>>>(o_bf, wo_bf, bo, aproj);
  k_ln<<<NT_, 256, 0, stream>>>(aproj, src, ln1_g, ln1_b, x_f, x_bf);

  // MoE
  k_gate<<<NT_ / 8, 256, 0, stream>>>(x_f, w_gate, cnt, elist, agate);
  k_moe_gemm1<<<dim3(DH_ / BN2, CAP_ / BM2, NE_), 256, 0, stream>>>(x_bf, w1t_bf, b1, cnt, elist, Hbuf);
  k_moe_gemm2<<<dim3(D_ / BN2, CAP_ / BM2, NE_), 256, 0, stream>>>(Hbuf, w2t_bf, b2, cnt, elist, agate, ff);

  k_ln<<<NT_, 256, 0, stream>>>(ff, x_f, ln2_g, ln2_b, (float*)d_out, nullptr);
}
